// LinearAttention_78915729097244
// MI455X (gfx1250) — matrix operand order repaired and checked by reading, not compiled, not run on hardware
//
#include <hip/hip_runtime.h>

// ---------------------------------------------------------------------------
// MI455X (gfx1250) LinearAttention block, bf16 WMMA pipeline.
// Memory-bound problem (~40 GFLOP vs ~1.3 GB traffic); all GEMMs use
// v_wmma_f32_16x16x32_bf16. The qkv GEMM stages its shared B tile into LDS
// with the Tensor Data Mover (tensor_load_to_lds + s_wait_tensorcnt), using
// TDM row padding for 32B-aligned, bank-staggered LDS rows. Other GEMMs load
// fragments directly from global in contiguous-k per-lane layouts (intermediates
// stored pre-transposed).
// ---------------------------------------------------------------------------

typedef __attribute__((ext_vector_type(16))) __bf16 v16bf;
typedef __attribute__((ext_vector_type(8)))  __bf16 v8bf;
typedef __attribute__((ext_vector_type(8)))  float  v8f;
typedef __attribute__((ext_vector_type(4)))  unsigned int u32x4;
typedef __attribute__((ext_vector_type(4)))  int i32x4;
typedef __attribute__((ext_vector_type(8)))  int i32x8;

#define B_      16
#define C_      128
#define HW_     16384      // 128*128
#define HEADS_  4
#define D_      32
#define GSIZE_  65536      // elements per (batch, group) = 4 channels * HW

// TDM-staged B tile for the qkv GEMM: 64 pixels x 128 ch bf16, padded by TDM
// with 8 DWORDs every 64 DWORDs -> row stride 144 elements (288 B, 32B align).
#define XN_TILE_ROWS   64
#define XN_LDS_STRIDE  144

__device__ __forceinline__ __bf16 f2bf(float f) {
  unsigned u = __builtin_bit_cast(unsigned, f);
  unsigned r = u + 0x7FFFu + ((u >> 16) & 1u);   // round-to-nearest-even
  unsigned short s = (unsigned short)(r >> 16);
  return __builtin_bit_cast(__bf16, s);
}
__device__ __forceinline__ float bf2f(__bf16 b) {
  unsigned short s = __builtin_bit_cast(unsigned short, b);
  unsigned u = ((unsigned)s) << 16;
  return __builtin_bit_cast(float, u);
}
__device__ __forceinline__ float elu1(float v) {   // elu(v)+1
  return v > 0.f ? v + 1.f : __expf(v);
}

// A-fragment (16x32 bf16, M x K): lane l -> row l%16, K = (l/16)*16+[0,16)
// (contiguous order, matching the B-fragment loads).
__device__ __forceinline__ v16bf load_a_frag(const __bf16* base, int stride, int lane) {
  int row  = lane & 15;
  int koff = (lane >> 4) * 16;
  const __bf16* r = base + (size_t)row * stride;
  v8bf lo = *(const v8bf*)(r + koff);
  v8bf hi = *(const v8bf*)(r + koff + 8);
  v16bf out;
  for (int i = 0; i < 8; ++i) { out[i] = lo[i]; out[8 + i] = hi[i]; }
  return out;
}

__device__ __forceinline__ v8f wmma_bf(v16bf a, v16bf b, v8f c) {
  return __builtin_amdgcn_wmma_f32_16x16x32_bf16(false, a, false, b, (short)0, c,
                                                 false, false);
}
__device__ __forceinline__ v8f zero8() {
  v8f z; for (int i = 0; i < 8; ++i) z[i] = 0.f; return z;
}

// ---------------------------------------------------------------------------
// K0: zero accumulators, convert weights fp32 -> bf16
// ---------------------------------------------------------------------------
__global__ void k_init(const float* __restrict__ wqkv, const float* __restrict__ wout,
                       __bf16* __restrict__ wq_bf, __bf16* __restrict__ wo_bf,
                       float* __restrict__ kv, float* __restrict__ ksum,
                       float* __restrict__ g2sum, float* __restrict__ g2sq) {
  int i = blockIdx.x * 256 + threadIdx.x;
  if (i < 384 * 128) wq_bf[i] = f2bf(wqkv[i]);
  if (i < 128 * 128) wo_bf[i] = f2bf(wout[i]);
  if (i < 64 * 1024) kv[i] = 0.f;
  if (i < 64 * 32)   ksum[i] = 0.f;
  if (i < 512)       { g2sum[i] = 0.f; g2sq[i] = 0.f; }
}

// ---------------------------------------------------------------------------
// K1: GroupNorm1 stats. Group's 4 channels are contiguous -> block streams
// 64K contiguous floats. stats[bg] = {mean, rstd}.
// ---------------------------------------------------------------------------
__global__ void k_gn1_stats(const float* __restrict__ x, float* __restrict__ stats) {
  int bg = blockIdx.x;                       // b*32 + g
  const float* base = x + (size_t)bg * GSIZE_;
  float s = 0.f, ss = 0.f;
  for (int i = threadIdx.x; i < GSIZE_ / 4; i += 256) {
    float4 v = ((const float4*)base)[i];
    s  += v.x + v.y + v.z + v.w;
    ss += v.x * v.x + v.y * v.y + v.z * v.z + v.w * v.w;
  }
  __shared__ float sh0[256], sh1[256];
  sh0[threadIdx.x] = s; sh1[threadIdx.x] = ss;
  __syncthreads();
  for (int st = 128; st > 0; st >>= 1) {
    if (threadIdx.x < st) {
      sh0[threadIdx.x] += sh0[threadIdx.x + st];
      sh1[threadIdx.x] += sh1[threadIdx.x + st];
    }
    __syncthreads();
  }
  if (threadIdx.x == 0) {
    float mean = sh0[0] * (1.f / GSIZE_);
    float var  = sh1[0] * (1.f / GSIZE_) - mean * mean;
    stats[bg * 2]     = mean;
    stats[bg * 2 + 1] = rsqrtf(var + 1e-5f);
  }
}

// ---------------------------------------------------------------------------
// K2: apply GN1, write xn bf16 TOKEN-MAJOR [b][p][c] (= transposed-B layout
// for the qkv GEMM). One thread per pixel; 16-B vector stores.
// ---------------------------------------------------------------------------
__global__ void k_gn1_apply(const float* __restrict__ x, const float* __restrict__ stats,
                            const float* __restrict__ gw, const float* __restrict__ gb,
                            __bf16* __restrict__ xn) {
  int b = blockIdx.y;
  int p = blockIdx.x * 256 + threadIdx.x;
  __shared__ float sw[128], sb[128], sm[32], sr[32];
  if (threadIdx.x < 128) { sw[threadIdx.x] = gw[threadIdx.x]; sb[threadIdx.x] = gb[threadIdx.x]; }
  if (threadIdx.x < 32) {
    sm[threadIdx.x] = stats[(b * 32 + threadIdx.x) * 2];
    sr[threadIdx.x] = stats[(b * 32 + threadIdx.x) * 2 + 1];
  }
  __syncthreads();
  const float* xb = x + (size_t)b * C_ * HW_ + p;
  __bf16* o = xn + ((size_t)b * HW_ + p) * C_;
  for (int c0 = 0; c0 < C_; c0 += 8) {
    v8bf pack;
    for (int j = 0; j < 8; ++j) {
      int c = c0 + j;
      float v = xb[(size_t)c * HW_];
      int g = c >> 2;
      pack[j] = f2bf((v - sm[g]) * sr[g] * sw[c] + sb[c]);
    }
    *(v8bf*)(o + c0) = pack;
  }
}

// ---------------------------------------------------------------------------
// K3: qkv = Wqkv[384x128] x Xn^T, per batch. Block tile 64(M)x64(N), 8 waves
// (4x2), wave tile 16x32, K = 128 in 4 WMMA steps. The 16 KB B tile (64
// pixels x 128 ch) is DMA'd into LDS once per block by the Tensor Data Mover
// (tensor_load_to_lds, TENSORcnt), with TDM row padding (8 DW per 64 DW) so
// rows are 32B-aligned and bank-staggered. Epilogue: elu+1 on q,k;
// q -> token-major [bh][n][d], k/v -> channel-major [bh][d][n].
// ---------------------------------------------------------------------------
__global__ void k_qkv_gemm(const __bf16* __restrict__ wq, const __bf16* __restrict__ xn,
                           __bf16* __restrict__ qbuf, __bf16* __restrict__ kbuf,
                           __bf16* __restrict__ vbuf) {
  __shared__ __bf16 smem[XN_TILE_ROWS * XN_LDS_STRIDE];
  const int lane = threadIdx.x & 31;
  const int wave = threadIdx.x >> 5;
  const int wm = wave >> 1, wn = wave & 1;
  const int b    = blockIdx.z;
  const int m0   = blockIdx.y * 64 + wm * 16;
  const int nblk = blockIdx.x * 64;
  const int n0   = nblk + wn * 32;

  if (wave == 0) {
    // ---- Tensor DMA descriptor (ISA 8.3/8.4): 2D tensor [16384][128] bf16,
    // tile [64][128] at row nblk, destination = smem with padding.
    unsigned long long gaddr =
        (unsigned long long)(uintptr_t)(xn + ((size_t)b * HW_ + nblk) * C_);
    unsigned lds_base = (unsigned)(uintptr_t)(&smem[0]);
    u32x4 g0;
    g0[0] = 1u;                                     // count=1 (valid user D#)
    g0[1] = lds_base;                               // lds_addr
    g0[2] = (unsigned)gaddr;                        // global_addr[31:0]
    g0[3] = (unsigned)(gaddr >> 32) | (2u << 30);   // global_addr[56:32] | type=2
    i32x8 g1;
    // data_size=1(2B) | pad_enable | pad_interval=5 (64 DW) | pad_amount=7 (8 DW)
    g1[0] = (int)0x0F510000u;
    g1[1] = (int)(C_ << 16);          // tensor_dim0 = 128       (bits 63:48)
    g1[2] = (int)(HW_ << 16);         // tensor_dim1 = 16384     (bits 95:80)
    g1[3] = (int)(C_ << 16);          // tile_dim0 = 128         (bits 127:112)
    g1[4] = XN_TILE_ROWS;             // tile_dim1 = 64          (bits 143:128)
    g1[5] = C_;                       // tensor_dim0_stride = 128 (bits 191:160)
    g1[6] = 0;
    g1[7] = 0;
    i32x4 gz; gz[0] = 0; gz[1] = 0; gz[2] = 0; gz[3] = 0;
#if defined(__clang_major__) && (__clang_major__ >= 23)
    i32x8 gz8; for (int i = 0; i < 8; ++i) gz8[i] = 0;
    __builtin_amdgcn_tensor_load_to_lds(g0, g1, gz, gz, gz8, 0);
#else
    __builtin_amdgcn_tensor_load_to_lds(g0, g1, gz, gz, 0);
#endif
    __builtin_amdgcn_s_wait_tensorcnt(0);
  }
  __syncthreads();

  v8f c0 = zero8(), c1 = zero8();
  for (int kk = 0; kk < 128; kk += 32) {
    v16bf a = load_a_frag(wq + m0 * 128 + kk, 128, lane);
    const __bf16* bp =
        smem + (wn * 32 + (lane & 15)) * XN_LDS_STRIDE + kk + ((lane >> 4) * 16);
    v16bf b0 = *(const v16bf*)bp;
    v16bf b1 = *(const v16bf*)(bp + 16 * XN_LDS_STRIDE);
    c0 = wmma_bf(a, b0, c0);
    c1 = wmma_bf(a, b1, c1);
  }
  const int rowadd = (lane >> 4) << 3;
  const int col    = lane & 15;
  for (int acc = 0; acc < 2; ++acc) {
    const v8f& cc = acc ? c1 : c0;
    int p = n0 + acc * 16 + col;
    for (int j = 0; j < 8; ++j) {
      float v = cc[j];
      int o = m0 + j + rowadd;
      if (o < 128) {                              // q: token-major
        int hd = o >> 5, d = o & 31;
        qbuf[(((size_t)b * HEADS_ + hd) * HW_ + p) * D_ + d] = f2bf(elu1(v));
      } else if (o < 256) {                       // k: channel-major
        int oo = o - 128;
        kbuf[(((size_t)b * HEADS_ + (oo >> 5)) * D_ + (oo & 31)) * (size_t)HW_ + p] =
            f2bf(elu1(v));
      } else {                                    // v: channel-major
        int oo = o - 256;
        vbuf[(((size_t)b * HEADS_ + (oo >> 5)) * D_ + (oo & 31)) * (size_t)HW_ + p] =
            f2bf(v);
      }
    }
  }
}

// ---------------------------------------------------------------------------
// K4: kv[d][e] = sum_n k[d][n] * v[e][n]  (per bh), plus ksum[d].
// Grid (64 bh, 8 n-chunks). Each wave covers 256 n (8 K-steps of 32),
// accumulating the full 32x32 tile (4 WMMA accumulators). Block reduce in
// LDS, then atomicAdd to global. ksum via per-lane VALU sums + ds_add_f32.
// ---------------------------------------------------------------------------
__global__ void k_kv_gemm(const __bf16* __restrict__ kbuf, const __bf16* __restrict__ vbuf,
                          float* __restrict__ kv, float* __restrict__ ksum) {
  const int lane = threadIdx.x & 31;
  const int wave = threadIdx.x >> 5;
  const int bh = blockIdx.x;
  const int chunk = blockIdx.y;
  const __bf16* kb = kbuf + (size_t)bh * D_ * HW_;
  const __bf16* vb = vbuf + (size_t)bh * D_ * HW_;
  __shared__ float lds_kv[8][1024];
  __shared__ float s_ks[32];
  if (threadIdx.x < 32) s_ks[threadIdx.x] = 0.f;
  __syncthreads();

  v8f acc[2][2];
  acc[0][0] = zero8(); acc[0][1] = zero8(); acc[1][0] = zero8(); acc[1][1] = zero8();
  float psum0 = 0.f, psum1 = 0.f;
  const int nstart = chunk * 2048 + wave * 256;
  for (int s = 0; s < 8; ++s) {
    int nk = nstart + s * 32;
    v16bf a0 = load_a_frag(kb + nk, HW_, lane);                       // d = 0..15
    v16bf a1 = load_a_frag(kb + (size_t)16 * HW_ + nk, HW_, lane);    // d = 16..31
    const __bf16* vp = vb + (size_t)(lane & 15) * HW_ + nk + ((lane >> 4) * 16);
    v16bf b0 = *(const v16bf*)vp;                                     // e = 0..15
    v16bf b1 = *(const v16bf*)(vp + (size_t)16 * HW_);                // e = 16..31
    acc[0][0] = wmma_bf(a0, b0, acc[0][0]);
    acc[0][1] = wmma_bf(a0, b1, acc[0][1]);
    acc[1][0] = wmma_bf(a1, b0, acc[1][0]);
    acc[1][1] = wmma_bf(a1, b1, acc[1][1]);
    for (int i = 0; i < 16; ++i) { psum0 += bf2f(a0[i]); psum1 += bf2f(a1[i]); }
  }
  const int rowadd = (lane >> 4) << 3;
  const int col    = lane & 15;
  for (int mt = 0; mt < 2; ++mt)
    for (int et = 0; et < 2; ++et)
      for (int j = 0; j < 8; ++j) {
        int m = mt * 16 + j + rowadd;
        int e = et * 16 + col;
        lds_kv[wave][m * 32 + e] = acc[mt][et][j];
      }
  atomicAdd(&s_ks[lane & 15], psum0);          // lanes l and l+16 hold the two
  atomicAdd(&s_ks[16 + (lane & 15)], psum1);   // K-halves of the same row
  __syncthreads();
  for (int idx = threadIdx.x; idx < 1024; idx += 256) {
    float s = 0.f;
    for (int w2 = 0; w2 < 8; ++w2) s += lds_kv[w2][idx];
    atomicAdd(&kv[(size_t)bh * 1024 + idx], s);
  }
  if (threadIdx.x < 32)
    atomicAdd(&ksum[(size_t)bh * 32 + threadIdx.x], s_ks[threadIdx.x]);
}

// ---------------------------------------------------------------------------
// K5: kv -> bf16 transposed [e][d]  (B-fragment layout for K6)
// ---------------------------------------------------------------------------
__global__ void k_kvT(const float* __restrict__ kv, __bf16* __restrict__ kvT) {
  int bh = blockIdx.x;
  for (int t = threadIdx.x; t < 1024; t += 256) {
    int d = t >> 5, e = t & 31;
    kvT[(size_t)bh * 1024 + e * 32 + d] = f2bf(kv[(size_t)bh * 1024 + d * 32 + e]);
  }
}

// ---------------------------------------------------------------------------
// K6: out_attn[n][e] = (sum_d q[n][d]*kv[d][e]) / (q.ksum + eps).
// Wave: 16 n-rows x 32 e, K = 32 -> single WMMA step per N-half, A fragment
// loaded straight from token-major q. Denominator via VALU dot + shfl_xor(16),
// broadcast through LDS. Output token-major [b][p][inner].
// ---------------------------------------------------------------------------
__global__ void k_attn(const __bf16* __restrict__ qbuf, const __bf16* __restrict__ kvT,
                       const float* __restrict__ ksum, __bf16* __restrict__ attn) {
  const int lane = threadIdx.x & 31;
  const int wave = threadIdx.x >> 5;
  const int bh = blockIdx.y;
  const int b  = bh >> 2, hd = bh & 3;
  const int n0 = blockIdx.x * 128 + wave * 16;
  __shared__ float sks[32];
  __shared__ float sden[8][16];
  if (threadIdx.x < 32) sks[threadIdx.x] = ksum[(size_t)bh * 32 + threadIdx.x];
  __syncthreads();

  const __bf16* qb = qbuf + ((size_t)bh * HW_ + n0) * D_;
  v16bf a = load_a_frag(qb, D_, lane);
  const __bf16* kp = kvT + (size_t)bh * 1024 + (size_t)(lane & 15) * D_ + ((lane >> 4) * 16);
  v16bf b0 = *(const v16bf*)kp;
  v16bf b1 = *(const v16bf*)(kp + 16 * D_);
  v8f c0 = zero8(), c1 = zero8();
  c0 = wmma_bf(a, b0, c0);
  c1 = wmma_bf(a, b1, c1);

  const int koff = (lane >> 4) * 16;
  float dot = 0.f;
  for (int i = 0; i < 8; ++i) {
    dot += bf2f(a[i])     * sks[koff + i];
    dot += bf2f(a[8 + i]) * sks[koff + 8 + i];
  }
  dot += __shfl_xor(dot, 16, 32);              // combine the two K-halves
  if (lane < 16) sden[wave][lane] = dot + 1e-6f;
  __syncthreads();

  const int rowadd = (lane >> 4) << 3;
  const int col    = lane & 15;
  for (int acc = 0; acc < 2; ++acc) {
    const v8f& cc = acc ? c1 : c0;
    for (int j = 0; j < 8; ++j) {
      int row = j + rowadd;                    // n-row within wave tile
      float v = cc[j] / sden[wave][row];
      int p = n0 + row;
      int e = acc * 16 + col;
      attn[((size_t)b * HW_ + p) * 128 + hd * 32 + e] = f2bf(v);
    }
  }
}

// ---------------------------------------------------------------------------
// K7: y = Wout[128x128] x attn^T, per batch; fused GroupNorm2 partial stats
// (LDS ds_add_f32 reduce -> 64 global atomics per block).
// ---------------------------------------------------------------------------
__global__ void k_out_gemm(const __bf16* __restrict__ wo, const __bf16* __restrict__ attn,
                           float* __restrict__ y, float* __restrict__ g2sum,
                           float* __restrict__ g2sq) {
  const int lane = threadIdx.x & 31;
  const int wave = threadIdx.x >> 5;
  const int wm = wave >> 1, wn = wave & 1;
  const int b  = blockIdx.z;
  const int m0 = blockIdx.y * 64 + wm * 16;
  const int n0 = blockIdx.x * 64 + wn * 32;
  __shared__ float ssum[32], ssq[32];
  if (threadIdx.x < 32) { ssum[threadIdx.x] = 0.f; ssq[threadIdx.x] = 0.f; }
  __syncthreads();

  const __bf16* ab = attn + (size_t)b * HW_ * 128;
  v8f c0 = zero8(), c1 = zero8();
  for (int kk = 0; kk < 128; kk += 32) {
    v16bf a = load_a_frag(wo + m0 * 128 + kk, 128, lane);
    const __bf16* bp = ab + (size_t)(n0 + (lane & 15)) * 128 + kk + ((lane >> 4) * 16);
    __builtin_prefetch(bp + 32, 0, 3);      // near-temporal prefetch of next K tile
    v16bf b0 = *(const v16bf*)bp;
    v16bf b1 = *(const v16bf*)(bp + 16 * 128);
    c0 = wmma_bf(a, b0, c0);
    c1 = wmma_bf(a, b1, c1);
  }
  const int rowadd = (lane >> 4) << 3;
  const int col    = lane & 15;
  for (int acc = 0; acc < 2; ++acc) {
    const v8f& cc = acc ? c1 : c0;
    int p = n0 + acc * 16 + col;
    for (int j = 0; j < 8; ++j) {
      float v = cc[j];
      int c = m0 + j + rowadd;
      y[((size_t)b * C_ + c) * HW_ + p] = v;
      atomicAdd(&ssum[c >> 2], v);
      atomicAdd(&ssq[c >> 2], v * v);
    }
  }
  __syncthreads();
  if (threadIdx.x < 32) {
    atomicAdd(&g2sum[b * 32 + threadIdx.x], ssum[threadIdx.x]);
    atomicAdd(&g2sq[b * 32 + threadIdx.x], ssq[threadIdx.x]);
  }
}

// ---------------------------------------------------------------------------
// K8: out = GN2(y) + residual, float4 vectorized
// ---------------------------------------------------------------------------
__global__ void k_final(const float* __restrict__ y, const float* __restrict__ x,
                        const float* __restrict__ g2sum, const float* __restrict__ g2sq,
                        const float* __restrict__ gw, const float* __restrict__ gb,
                        float* __restrict__ out) {
  size_t idx = (size_t)blockIdx.x * blockDim.x + threadIdx.x;  // float4 index
  size_t i = idx * 4;
  int b = (int)(i / ((size_t)C_ * HW_));
  int rem = (int)(i - (size_t)b * C_ * HW_);
  int c = rem / HW_;
  int g = c >> 2;
  float mean = g2sum[b * 32 + g] * (1.f / GSIZE_);
  float var  = g2sq[b * 32 + g] * (1.f / GSIZE_) - mean * mean;
  float rs   = rsqrtf(var + 1e-5f);
  float w = gw[c], bb = gb[c];
  float4 yv = ((const float4*)y)[idx];
  float4 xv = ((const float4*)x)[idx];
  float4 o;
  o.x = (yv.x - mean) * rs * w + bb + xv.x;
  o.y = (yv.y - mean) * rs * w + bb + xv.y;
  o.z = (yv.z - mean) * rs * w + bb + xv.z;
  o.w = (yv.w - mean) * rs * w + bb + xv.w;
  ((float4*)out)[idx] = o;
}

// ---------------------------------------------------------------------------
extern "C" void kernel_launch(void* const* d_in, const int* in_sizes, int n_in,
                              void* d_out, int out_size, void* d_ws, size_t ws_size,
                              hipStream_t stream) {
  (void)in_sizes; (void)n_in; (void)out_size; (void)ws_size;
  const float* x    = (const float*)d_in[0];
  const float* gn1w = (const float*)d_in[1];
  const float* gn1b = (const float*)d_in[2];
  const float* wqkv = (const float*)d_in[3];
  const float* wout = (const float*)d_in[4];
  const float* gn2w = (const float*)d_in[5];
  const float* gn2b = (const float*)d_in[6];
  float* out = (float*)d_out;

  char* ws = (char*)d_ws;
  size_t off = 0;
  auto alloc = [&](size_t bytes) -> char* {
    char* p = ws + off;
    off += (bytes + 255) & ~(size_t)255;
    return p;
  };
  float*  stats1 = (float*)alloc(512 * 2 * sizeof(float));
  float*  g2sum  = (float*)alloc(512 * sizeof(float));
  float*  g2sq   = (float*)alloc(512 * sizeof(float));
  float*  kv     = (float*)alloc(64 * 1024 * sizeof(float));
  float*  ksum   = (float*)alloc(64 * 32 * sizeof(float));
  __bf16* kvT    = (__bf16*)alloc(64 * 1024 * sizeof(__bf16));
  __bf16* wq_bf  = (__bf16*)alloc(384 * 128 * sizeof(__bf16));
  __bf16* wo_bf  = (__bf16*)alloc(128 * 128 * sizeof(__bf16));
  __bf16* xn     = (__bf16*)alloc((size_t)B_ * HW_ * C_ * sizeof(__bf16));       // 67 MB
  __bf16* qbuf   = (__bf16*)alloc((size_t)B_ * HEADS_ * HW_ * D_ * sizeof(__bf16));
  __bf16* kbuf   = (__bf16*)alloc((size_t)B_ * HEADS_ * HW_ * D_ * sizeof(__bf16));
  __bf16* vbuf   = (__bf16*)alloc((size_t)B_ * HEADS_ * HW_ * D_ * sizeof(__bf16));
  __bf16* attnb  = (__bf16*)alloc((size_t)B_ * HW_ * 128 * sizeof(__bf16));
  float*  ybuf   = (float*)alloc((size_t)B_ * C_ * HW_ * sizeof(float));         // 134 MB

  k_init<<<192, 256, 0, stream>>>(wqkv, wout, wq_bf, wo_bf, kv, ksum, g2sum, g2sq);
  k_gn1_stats<<<512, 256, 0, stream>>>(x, stats1);
  k_gn1_apply<<<dim3(HW_ / 256, B_), 256, 0, stream>>>(x, stats1, gn1w, gn1b, xn);
  k_qkv_gemm<<<dim3(HW_ / 64, 6, B_), 256, 0, stream>>>(wq_bf, xn, qbuf, kbuf, vbuf);
  k_kv_gemm<<<dim3(64, 8), 256, 0, stream>>>(kbuf, vbuf, kv, ksum);
  k_kvT<<<64, 256, 0, stream>>>(kv, kvT);
  k_attn<<<dim3(HW_ / 128, 64), 256, 0, stream>>>(qbuf, kvT, ksum, attnb);
  k_out_gemm<<<dim3(HW_ / 64, 2, B_), 256, 0, stream>>>(wo_bf, attnb, ybuf, g2sum, g2sq);
  k_final<<<(int)(((size_t)B_ * C_ * HW_ / 4) / 256), 256, 0, stream>>>(
      ybuf, x, g2sum, g2sq, gn2w, gn2b, out);
}
